// cGRULayer_63539746177422
// MI455X (gfx1250) — matrix operand order repaired and checked by reading, not compiled, not run on hardware
//
#include <hip/hip_runtime.h>
#include <math.h>

// ---- problem dims ----
#define T_STEPS 256
#define BATCH   64
#define DIN     512
#define DH      1024
#define DCTX    512
#define DA      256
#define DM      50
#define G3H     3072           // 3*H
#define KBIG    2048           // DIN + DH + DCTX
#define NBIG    4096           // 2H (r,z) + H (n from x,c) + H (n from h)

#define CHUNK_K    128         // K elements staged per TDM chunk
#define LDS_STRIDE 136         // 128 + 8 bf16 pad -> 272B row stride, 4-bank rotation

typedef __attribute__((ext_vector_type(16))) __bf16 v16bf;
typedef __attribute__((ext_vector_type(8)))  __bf16 v8bf;
typedef __attribute__((ext_vector_type(8)))  float  v8f;
typedef unsigned int v4u __attribute__((ext_vector_type(4)));
typedef int          v8i __attribute__((ext_vector_type(8)));
typedef int          v4i __attribute__((ext_vector_type(4)));

// TDM availability / arity detection (device pass only; host pass parses fallback)
#if defined(__HIP_DEVICE_COMPILE__) && __has_builtin(__builtin_amdgcn_tensor_load_to_lds)
#define HAVE_TDM 1
#else
#define HAVE_TDM 0
#endif
#if __has_include(<hip/amd_detail/amd_gfx1250_TDM.h>) || (__clang_major__ >= 23)
#define TDM_6ARG 1
#else
#define TDM_6ARG 0
#endif

// --- compile-time probe: these warnings surface in hipcc stderr so the
// compile-only loop can verify which path the DEVICE pass actually took. ---
#if HAVE_TDM
#if TDM_6ARG
#warning "CDNA5-PROBE: device pass using tensor_load_to_lds (6-arg form)"
#else
#warning "CDNA5-PROBE: device pass using tensor_load_to_lds (5-arg form)"
#endif
#else
#warning "CDNA5-PROBE: TDM builtin not visible in this pass (fallback copy path)"
#endif

// fp32 -> bf16 round-to-nearest-even, stored as raw u16 (avoids __bf16 in host code)
__device__ inline unsigned short f2bf(float f) {
    union { float f; unsigned u; } x; x.f = f;
    unsigned r = (x.u + 0x7fffu + ((x.u >> 16) & 1u)) >> 16;
    return (unsigned short)r;
}

__device__ inline v16bf join8(v8bf lo, v8bf hi) {
    return __builtin_shufflevector(lo, hi, 0,1,2,3,4,5,6,7,8,9,10,11,12,13,14,15);
}

// A fragment (16x32 bf16, row-major source, stride lda elements).
// lanes 0-15: row M=lane,   K = kb+[0..7]  and kb+[16..23]
// lanes16-31: row M=lane-16,K = kb+[8..15] and kb+[24..31]
__device__ inline v16bf load_afrag(const unsigned short* A, int lda, int rbase, int kb, int lane) {
    int half = lane >> 4;
    const unsigned short* p = A + (size_t)(rbase + (lane & 15)) * lda + kb + half * 8;
    v8bf a0 = *(const v8bf*)(p);
    v8bf a1 = *(const v8bf*)(p + 16);
    return join8(a0, a1);
}

// A fragment from padded LDS tile (row stride LDS_STRIDE elements)
__device__ inline v16bf load_afrag_lds(const unsigned short* base, int rbase, int kb, int lane) {
    int half = lane >> 4;
    const unsigned short* p = base + (rbase + (lane & 15)) * LDS_STRIDE + kb + half * 8;
    v8bf a0 = *(const v8bf*)(p);
    v8bf a1 = *(const v8bf*)(p + 16);
    return join8(a0, a1);
}

// B fragment (32x16 bf16) from pre-transposed weight Bt[N][K] (stride ldb).
// lanes 0-15: col N=nbase+lane,   K = kb+[0..7]  and kb+[16..23]
// lanes16-31: col N=nbase+lane-16,K = kb+[8..15] and kb+[24..31]
__device__ inline v16bf load_bfrag(const unsigned short* Bt, int ldb, int nbase, int kb, int lane) {
    int half = lane >> 4;
    const unsigned short* p = Bt + (size_t)(nbase + (lane & 15)) * ldb + kb + half * 8;
    v8bf b0 = *(const v8bf*)(p);
    v8bf b1 = *(const v8bf*)(p + 16);
    return join8(b0, b1);
}

// C/D f32 16x16: VGPR i holds M = rbase + i + 8*half, N = nbase + (lane&15)
__device__ inline void store_tile(float* O, int ldo, int rbase, int nbase, const v8f& acc, int lane) {
    int half = lane >> 4;
    int col  = nbase + (lane & 15);
#pragma unroll
    for (int i = 0; i < 8; ++i)
        O[(size_t)(rbase + half * 8 + i) * ldo + col] = acc[i];
}

// one wave accumulates a 16x64 strip (4 N-tiles) over klen (multiple of 32), A from global
__device__ inline void gemm_1x4_accum(const unsigned short* Aseg, int lda, int rbase,
                                      const unsigned short* Bseg, int ldb, int nbase,
                                      int klen, v8f acc[4], int lane) {
#pragma unroll 4
    for (int kb = 0; kb < klen; kb += 32) {
        v16bf af = load_afrag(Aseg, lda, rbase, kb, lane);
        v16bf b0 = load_bfrag(Bseg, ldb, nbase +  0, kb, lane);
        v16bf b1 = load_bfrag(Bseg, ldb, nbase + 16, kb, lane);
        v16bf b2 = load_bfrag(Bseg, ldb, nbase + 32, kb, lane);
        v16bf b3 = load_bfrag(Bseg, ldb, nbase + 48, kb, lane);
        acc[0] = __builtin_amdgcn_wmma_f32_16x16x32_bf16(false, af, false, b0, (short)0, acc[0], false, false);
        acc[1] = __builtin_amdgcn_wmma_f32_16x16x32_bf16(false, af, false, b1, (short)0, acc[1], false, false);
        acc[2] = __builtin_amdgcn_wmma_f32_16x16x32_bf16(false, af, false, b2, (short)0, acc[2], false, false);
        acc[3] = __builtin_amdgcn_wmma_f32_16x16x32_bf16(false, af, false, b3, (short)0, acc[3], false, false);
    }
}

#if HAVE_TDM
// TDM: async-copy a [64 rows x CHUNK_K] bf16 tile (row stride lda elements) from global
// into LDS at byte offset lds_off, inserting 16B padding after every 256B row so the
// LDS row stride becomes LDS_STRIDE*2 bytes. D# per CDNA5 ISA ch.8 (2D, data_size=2B).
__device__ inline void tdm_load_a_chunk(const unsigned short* src, int lda, unsigned lds_off) {
    unsigned long long ga = (unsigned long long)(uintptr_t)src;
    v4u g0 = { 1u,                                     // count=1, user mode
               lds_off,                                // lds_addr (bytes)
               (unsigned)ga,                           // global_addr[31:0]
               ((unsigned)(ga >> 32) & 0x1FFFFFFu) | (2u << 30) };  // addr[56:32] | type=2
    v8i g1 = { (int)((1u << 16)                        // data_size = 2 bytes
                   | (1u << 20)                        // pad_enable
                   | (5u << 22)                        // pad_interval: 2^(5+1)=64 dwords (256B)
                   | (3u << 25)),                      // pad_amount: 3+1=4 dwords (16B)
               (int)(((unsigned)lda & 0xFFFFu) << 16), // tensor_dim0 = lda (low 16)
               (int)((((unsigned)lda >> 16) & 0xFFFFu) | (64u << 16)), // dim0 hi | tensor_dim1=64
               (int)((unsigned)CHUNK_K << 16),         // tensor_dim1 hi=0 | tile_dim0=128
               (int)64,                                // tile_dim1=64 | tile_dim2=0
               (int)lda,                               // tensor_dim0_stride (data_size units)
               0, 0 };
    v4i z4 = {0, 0, 0, 0};
#if TDM_6ARG
    v8i z8 = {0, 0, 0, 0, 0, 0, 0, 0};
    __builtin_amdgcn_tensor_load_to_lds(g0, g1, z4, z4, z8, 0);
#else
    __builtin_amdgcn_tensor_load_to_lds(g0, g1, z4, z4, 0);
#endif
}
#endif

// ---------------- setup kernels ----------------

__global__ __launch_bounds__(256) void k_cvt_bf16(const float* __restrict__ src,
                                                  unsigned short* __restrict__ dst, int n) {
    int i = blockIdx.x * 256 + threadIdx.x;
    if (i < n) dst[i] = f2bf(src[i]);
}

// src f32 [K,N] row-major -> dst bf16 [N,K] row-major (transpose-convert)
__global__ __launch_bounds__(256) void k_tr_bf16(const float* __restrict__ src,
                                                 unsigned short* __restrict__ dst, int K, int N) {
    int i = blockIdx.x * 256 + threadIdx.x;
    if (i >= K * N) return;
    int n = i / K, k = i % K;
    dst[i] = f2bf(src[(size_t)k * N + n]);
}

// Wbig[g][k] bf16, g in [0,4096), k in [0,2048):
//  g <  2048 : r,z gate cols g        (x,h,c rows live)
//  g <  3072 : n gate cols g          (x,c rows live; h rows zero/never read)
//  g >= 3072 : n gate cols g-1024     (h rows only)
__global__ __launch_bounds__(256) void k_build_wbig(const float* __restrict__ Wx,
                                                    const float* __restrict__ Wh,
                                                    const float* __restrict__ Wc,
                                                    unsigned short* __restrict__ Wbig) {
    int i = blockIdx.x * 256 + threadIdx.x;            // < 4096*2048
    int g = i / KBIG, k = i % KBIG;
    float val = 0.f;
    if (k < DIN) {
        if (g < 3072) val = Wx[(size_t)k * G3H + g];
    } else if (k < DIN + DH) {
        int kk = k - DIN;
        if (g < 2048)       val = Wh[(size_t)kk * G3H + g];
        else if (g >= 3072) val = Wh[(size_t)kk * G3H + (g - 1024)];
    } else {
        int kk = k - DIN - DH;
        if (g < 3072) val = Wc[(size_t)kk * G3H + g];
    }
    Wbig[i] = f2bf(val);
}

__global__ __launch_bounds__(256) void k_init_h(const float* __restrict__ h0,
                                                float* __restrict__ h_f32,
                                                unsigned short* __restrict__ h_bf) {
    int i = blockIdx.x * 256 + threadIdx.x;            // < 64*1024
    float v = h0[i];
    h_f32[i] = v;
    h_bf[i]  = f2bf(v);
}

// ctx_proj[B*M, A] = ctx_bf[B*M, CTX] @ Wac_t^T  (Wac_t is [A, CTX] bf16)
__global__ __launch_bounds__(256) void k_ctxproj(const unsigned short* __restrict__ ctx_bf,
                                                 const unsigned short* __restrict__ Wac_t,
                                                 float* __restrict__ ctx_proj) {
    int wid  = (blockIdx.x * 256 + threadIdx.x) >> 5;  // 800 waves
    int lane = threadIdx.x & 31;
    int mtile = wid >> 2;                              // 0..199
    int ng    = wid & 3;                               // 0..3
    int rbase = mtile * 16, nbase = ng * 64;
    v8f acc[4] = {};
    gemm_1x4_accum(ctx_bf, DCTX, rbase, Wac_t, DCTX, nbase, DCTX, acc, lane);
#pragma unroll
    for (int j = 0; j < 4; ++j) store_tile(ctx_proj, DA, rbase, nbase + j * 16, acc[j], lane);
}

// ---------------- per-step kernels ----------------

// hWa[64, 256] = h_bf[64,1024] @ Wa_h   (Wah_t is [A, H] bf16)
__global__ __launch_bounds__(256) void k_att1(const unsigned short* __restrict__ h_bf,
                                              const unsigned short* __restrict__ Wah_t,
                                              float* __restrict__ hWa) {
    int wid  = (blockIdx.x * 256 + threadIdx.x) >> 5;  // 16 waves
    int lane = threadIdx.x & 31;
    int mtile = wid >> 2, ng = wid & 3;
    int rbase = mtile * 16, nbase = ng * 64;
    v8f acc[4] = {};
    gemm_1x4_accum(h_bf, DH, rbase, Wah_t, DH, nbase, DH, acc, lane);
#pragma unroll
    for (int j = 0; j < 4; ++j) store_tile(hWa, DA, rbase, nbase + j * 16, acc[j], lane);
}

// attention scores + softmax + context vector; one block per batch element
__global__ __launch_bounds__(256) void k_att2(const float* __restrict__ ctx_proj,
                                              const float* __restrict__ hWa,
                                              const float* __restrict__ vvec,
                                              const float* __restrict__ ctx,
                                              float* __restrict__ cs_out,
                                              unsigned short* __restrict__ c_bf) {
    int b = blockIdx.x, tid = threadIdx.x;
    int lane = tid & 31, w = tid >> 5;
    __shared__ float e_sm[56];
    __shared__ float alpha_sm[56];
    __shared__ float red[2];

    const float* hp = hWa + (size_t)b * DA;
    for (int m = w; m < DM; m += 8) {
        const float* cp = ctx_proj + ((size_t)b * DM + m) * DA;
        float s = 0.f;
        for (int a = lane; a < DA; a += 32)
            s += vvec[a] * tanhf(cp[a] + hp[a]);
        for (int off = 16; off; off >>= 1) s += __shfl_xor(s, off, 32);
        if (lane == 0) e_sm[m] = s;
    }
    __syncthreads();
    if (tid == 0) {
        float mx = -1e30f;
        for (int m = 0; m < DM; ++m) mx = fmaxf(mx, e_sm[m]);
        float den = 0.f;
        for (int m = 0; m < DM; ++m) den += expf(e_sm[m] - mx);
        red[0] = mx; red[1] = den;
    }
    __syncthreads();
    if (tid < DM) alpha_sm[tid] = expf(e_sm[tid] - red[0]) / red[1];
    __syncthreads();
    for (int x = tid; x < DCTX; x += 256) {
        const float* cb = ctx + (size_t)b * DM * DCTX + x;
        float acc = 0.f;
        for (int m = 0; m < DM; ++m) acc += alpha_sm[m] * cb[(size_t)m * DCTX];
        cs_out[(size_t)b * DCTX + x] = acc;
        c_bf[(size_t)b * DCTX + x]   = f2bf(acc);
    }
}

// O[64, 4096] = [x_t | h | c] @ Wbig.
// A activations staged into LDS by the Tensor Data Mover (double-buffered CHUNK_K tiles,
// TDM row padding -> conflict-free ds_load A fragments); B (weights) streams from L2.
__global__ __launch_bounds__(256) void k_gate_gemm(const unsigned short* __restrict__ xbf_t,
                                                   const unsigned short* __restrict__ h_bf,
                                                   const unsigned short* __restrict__ c_bf,
                                                   const unsigned short* __restrict__ Wbig,
                                                   float* __restrict__ O) {
    __shared__ unsigned short lds_a[2][64 * LDS_STRIDE];   // 2 x 17 KB

    int tid  = threadIdx.x;
    int wid  = (blockIdx.x * 256 + tid) >> 5;          // 256 waves total
    int lane = tid & 31;
    int mtile = wid & 3;                               // 0..3
    int ng    = wid >> 2;                              // 0..63
    int rbase = mtile * 16, nbase = ng * 64;

    // K segments for this N-region
    const unsigned short* sbase[3];
    int slda[3], skg[3], sklen[3], nseg;
    if (nbase < 2048) {                                // r,z gates: full K
        sbase[0] = xbf_t; slda[0] = DIN;  skg[0] = 0;         sklen[0] = DIN;
        sbase[1] = h_bf;  slda[1] = DH;   skg[1] = DIN;       sklen[1] = DH;
        sbase[2] = c_bf;  slda[2] = DCTX; skg[2] = DIN + DH;  sklen[2] = DCTX;
        nseg = 3;
    } else if (nbase < 3072) {                         // n gate: x + c
        sbase[0] = xbf_t; slda[0] = DIN;  skg[0] = 0;         sklen[0] = DIN;
        sbase[1] = c_bf;  slda[1] = DCTX; skg[1] = DIN + DH;  sklen[1] = DCTX;
        nseg = 2;
    } else {                                           // n gate: h only
        sbase[0] = h_bf;  slda[0] = DH;   skg[0] = DIN;       sklen[0] = DH;
        nseg = 1;
    }

    v8f acc[4] = {};
#if HAVE_TDM
    bool issuer = (tid < 32);                          // one wave drives the TDM
    unsigned lds_off0 = (unsigned)(uintptr_t)&lds_a[0][0];
    unsigned lds_off1 = (unsigned)(uintptr_t)&lds_a[1][0];
#endif

    for (int s = 0; s < nseg; ++s) {
        int nch = sklen[s] / CHUNK_K;
#if HAVE_TDM
        if (issuer) tdm_load_a_chunk(sbase[s], slda[s], lds_off0);
        for (int ci = 0; ci < nch; ++ci) {
            int cur = ci & 1;
            if (issuer) {
                if (ci + 1 < nch) {
                    tdm_load_a_chunk(sbase[s] + (ci + 1) * CHUNK_K, slda[s],
                                     cur ? lds_off0 : lds_off1);
                    __builtin_amdgcn_s_wait_tensorcnt(1);  // chunk ci has landed
                } else {
                    __builtin_amdgcn_s_wait_tensorcnt(0);
                }
            }
            __syncthreads();
            const unsigned short* abase = &lds_a[cur][0];
            int kg0 = skg[s] + ci * CHUNK_K;
#pragma unroll
            for (int kb = 0; kb < CHUNK_K; kb += 32) {
                v16bf af = load_afrag_lds(abase, rbase, kb, lane);
                int kg = kg0 + kb;
                v16bf b0 = load_bfrag(Wbig, KBIG, nbase +  0, kg, lane);
                v16bf b1 = load_bfrag(Wbig, KBIG, nbase + 16, kg, lane);
                v16bf b2 = load_bfrag(Wbig, KBIG, nbase + 32, kg, lane);
                v16bf b3 = load_bfrag(Wbig, KBIG, nbase + 48, kg, lane);
                acc[0] = __builtin_amdgcn_wmma_f32_16x16x32_bf16(false, af, false, b0, (short)0, acc[0], false, false);
                acc[1] = __builtin_amdgcn_wmma_f32_16x16x32_bf16(false, af, false, b1, (short)0, acc[1], false, false);
                acc[2] = __builtin_amdgcn_wmma_f32_16x16x32_bf16(false, af, false, b2, (short)0, acc[2], false, false);
                acc[3] = __builtin_amdgcn_wmma_f32_16x16x32_bf16(false, af, false, b3, (short)0, acc[3], false, false);
            }
            __syncthreads();                           // buffer reuse guard
        }
#else
        // fallback: cooperative copy through normal loads (also used by host pass)
        for (int ci = 0; ci < nch; ++ci) {
            for (int e = tid; e < 64 * (CHUNK_K / 2); e += 256) {
                int row = e / (CHUNK_K / 2), col = e % (CHUNK_K / 2);
                const unsigned* srcp =
                    (const unsigned*)(sbase[s] + (size_t)row * slda[s] + ci * CHUNK_K) + col;
                ((unsigned*)&lds_a[0][0])[row * (LDS_STRIDE / 2) + col] = *srcp;
            }
            __syncthreads();
            const unsigned short* abase = &lds_a[0][0];
            int kg0 = skg[s] + ci * CHUNK_K;
#pragma unroll
            for (int kb = 0; kb < CHUNK_K; kb += 32) {
                v16bf af = load_afrag_lds(abase, rbase, kb, lane);
                int kg = kg0 + kb;
                v16bf b0 = load_bfrag(Wbig, KBIG, nbase +  0, kg, lane);
                v16bf b1 = load_bfrag(Wbig, KBIG, nbase + 16, kg, lane);
                v16bf b2 = load_bfrag(Wbig, KBIG, nbase + 32, kg, lane);
                v16bf b3 = load_bfrag(Wbig, KBIG, nbase + 48, kg, lane);
                acc[0] = __builtin_amdgcn_wmma_f32_16x16x32_bf16(false, af, false, b0, (short)0, acc[0], false, false);
                acc[1] = __builtin_amdgcn_wmma_f32_16x16x32_bf16(false, af, false, b1, (short)0, acc[1], false, false);
                acc[2] = __builtin_amdgcn_wmma_f32_16x16x32_bf16(false, af, false, b2, (short)0, acc[2], false, false);
                acc[3] = __builtin_amdgcn_wmma_f32_16x16x32_bf16(false, af, false, b3, (short)0, acc[3], false, false);
            }
            __syncthreads();
        }
#endif
    }

#pragma unroll
    for (int j = 0; j < 4; ++j) store_tile(O, NBIG, rbase, nbase + j * 16, acc[j], lane);
}

// gate nonlinearity + hidden-state update
__global__ __launch_bounds__(256) void k_gate2(const float* __restrict__ O,
                                               const float* __restrict__ bias,
                                               float* __restrict__ h_f32,
                                               unsigned short* __restrict__ h_bf,
                                               float* __restrict__ hs_out) {
    int idx = blockIdx.x * 256 + threadIdx.x;          // < 64*1024
    int b = idx >> 10, j = idx & 1023;
    const float* Ob = O + (size_t)b * NBIG;
    float r = 1.f / (1.f + expf(-(Ob[j]        + bias[j])));
    float z = 1.f / (1.f + expf(-(Ob[1024 + j] + bias[1024 + j])));
    float n = tanhf(Ob[2048 + j] + bias[2048 + j] + r * Ob[3072 + j]);
    float h = (1.f - z) * n + z * h_f32[idx];
    h_f32[idx] = h;
    h_bf[idx]  = f2bf(h);
    hs_out[idx] = h;
}

// ---------------- host launcher ----------------

extern "C" void kernel_launch(void* const* d_in, const int* in_sizes, int n_in,
                              void* d_out, int out_size, void* d_ws, size_t ws_size,
                              hipStream_t stream) {
    const float* x_seq = (const float*)d_in[0];   // [T,B,IN]
    const float* ctx   = (const float*)d_in[1];   // [B,M,CTX]
    const float* h0    = (const float*)d_in[2];   // [B,H]
    const float* Wx    = (const float*)d_in[3];   // [IN,3H]
    const float* Wh    = (const float*)d_in[4];   // [H,3H]
    const float* Wc    = (const float*)d_in[5];   // [CTX,3H]
    const float* bias  = (const float*)d_in[6];   // [3H]
    const float* Wah   = (const float*)d_in[7];   // [H,A]
    const float* Wac   = (const float*)d_in[8];   // [CTX,A]
    const float* vvec  = (const float*)d_in[9];   // [A]

    float* cs_base = (float*)d_out;                                   // [T,B,CTX]
    float* hs_base = (float*)d_out + (size_t)T_STEPS * BATCH * DCTX;  // [T,B,H]

    // workspace carve-out (256B aligned)
    char* ws = (char*)d_ws;
    size_t off = 0;
    auto alloc = [&](size_t bytes) { size_t o = off; off = (off + bytes + 255) & ~(size_t)255; return o; };
    unsigned short* x_bf    = (unsigned short*)(ws + alloc((size_t)T_STEPS * BATCH * DIN * 2));
    unsigned short* ctx_bf  = (unsigned short*)(ws + alloc((size_t)BATCH * DM * DCTX * 2));
    unsigned short* Wbig    = (unsigned short*)(ws + alloc((size_t)NBIG * KBIG * 2));
    unsigned short* Wah_t   = (unsigned short*)(ws + alloc((size_t)DA * DH * 2));
    unsigned short* Wac_t   = (unsigned short*)(ws + alloc((size_t)DA * DCTX * 2));
    float*          ctxproj = (float*)         (ws + alloc((size_t)BATCH * DM * DA * 4));
    float*          h_f32   = (float*)         (ws + alloc((size_t)BATCH * DH * 4));
    unsigned short* h_bf    = (unsigned short*)(ws + alloc((size_t)BATCH * DH * 2));
    unsigned short* c_bf    = (unsigned short*)(ws + alloc((size_t)BATCH * DCTX * 2));
    float*          hWa     = (float*)         (ws + alloc((size_t)BATCH * DA * 4));
    float*          Obuf    = (float*)         (ws + alloc((size_t)BATCH * NBIG * 4));

    // ---- setup ----
    {
        int n = T_STEPS * BATCH * DIN;
        k_cvt_bf16<<<(n + 255) / 256, 256, 0, stream>>>(x_seq, x_bf, n);
    }
    {
        int n = BATCH * DM * DCTX;
        k_cvt_bf16<<<(n + 255) / 256, 256, 0, stream>>>(ctx, ctx_bf, n);
    }
    k_tr_bf16<<<(DH * DA + 255) / 256, 256, 0, stream>>>(Wah, Wah_t, DH, DA);
    k_tr_bf16<<<(DCTX * DA + 255) / 256, 256, 0, stream>>>(Wac, Wac_t, DCTX, DA);
    k_build_wbig<<<(NBIG * KBIG) / 256, 256, 0, stream>>>(Wx, Wh, Wc, Wbig);
    k_init_h<<<(BATCH * DH) / 256, 256, 0, stream>>>(h0, h_f32, h_bf);
    k_ctxproj<<<100, 256, 0, stream>>>(ctx_bf, Wac_t, ctxproj);   // 800 waves

    // ---- sequential recurrence: 4 kernels per timestep ----
    for (int t = 0; t < T_STEPS; ++t) {
        float* cs_t = cs_base + (size_t)t * BATCH * DCTX;
        float* hs_t = hs_base + (size_t)t * BATCH * DH;
        const unsigned short* xbf_t = x_bf + (size_t)t * BATCH * DIN;

        k_att1<<<2, 256, 0, stream>>>(h_bf, Wah_t, hWa);                     // 16 waves
        k_att2<<<BATCH, 256, 0, stream>>>(ctxproj, hWa, vvec, ctx, cs_t, c_bf);
        k_gate_gemm<<<32, 256, 0, stream>>>(xbf_t, h_bf, c_bf, Wbig, Obuf);  // 256 waves
        k_gate2<<<(BATCH * DH) / 256, 256, 0, stream>>>(Obuf, bias, h_f32, h_bf, hs_t);
    }
}
